// Head_56023553409128
// MI455X (gfx1250) — matrix operand order repaired and checked by reading, not compiled, not run on hardware
//
#include <hip/hip_runtime.h>
#include <hip/hip_bf16.h>
#include <stdint.h>

// ---------------------------------------------------------------------------
// Problem constants (B=4, T=2048, C=2048, H=128)
// ---------------------------------------------------------------------------
#define BB   4
#define TT   2048
#define CC   2048
#define HH   128
#define BT   (BB * TT)          // 8192 rows total

typedef __attribute__((ext_vector_type(16))) __bf16 v16bf;
typedef __attribute__((ext_vector_type(8)))  float  v8f;

union FragBF {
  v16bf v;
  uint4 q[2];
  unsigned short us[16];
};

// Native f32 -> bf16 converts (v_cvt_bf16_f32 / v_cvt_pk_bf16_f32)
__device__ __forceinline__ unsigned short bf1(float a) {
  union { __bf16 h; unsigned short u; } t;
  t.h = (__bf16)a;
  return t.u;
}
__device__ __forceinline__ unsigned pk2bf(float a, float b) {
  union { __bf16 h[2]; unsigned u; } t;
  t.h[0] = (__bf16)a;
  t.h[1] = (__bf16)b;
  return t.u;
}

// ---------------------------------------------------------------------------
// Kernel 0: pack Wq/Wk/Wv (fp32 [C,H]) into bf16 WMMA B-fragment layout.
// Tile (m, kt, nt) = [kt*32 .. +32) x [nt*16 .. +16).  Per ISA B layout:
// lane n in 0..15 holds K=0..7,16..23 of column n, lane n+16 holds K=8..15,24..31.
// Stored as [tile][lane][16 bf16] so the GEMM loads are two b128 per lane.
// ---------------------------------------------------------------------------
__global__ __launch_bounds__(256)
void pack_w_kernel(const float* __restrict__ Wq,
                   const float* __restrict__ Wk,
                   const float* __restrict__ Wv,
                   unsigned short* __restrict__ Wp) {
  const int wave = (blockIdx.x * blockDim.x + threadIdx.x) >> 5;  // 0..1535
  const int lane = threadIdx.x & 31;
  const int m  = wave >> 9;          // 0..2  (which matrix)
  const int kt = (wave >> 3) & 63;   // K tile (32 wide)
  const int nt = wave & 7;           // N tile (16 wide)
  const float* W = (m == 0) ? Wq : (m == 1) ? Wk : Wv;
  const int n  = lane & 15;
  const int kh = lane >> 4;

  float f[16];
#pragma unroll
  for (int i = 0; i < 16; ++i) {
    int kk = kt * 32 + (i < 8 ? kh * 8 + i : 16 + kh * 8 + (i - 8));
    f[i] = W[(size_t)kk * HH + nt * 16 + n];
  }
  uint4 v0, v1;
  v0.x = pk2bf(f[0],  f[1]);
  v0.y = pk2bf(f[2],  f[3]);
  v0.z = pk2bf(f[4],  f[5]);
  v0.w = pk2bf(f[6],  f[7]);
  v1.x = pk2bf(f[8],  f[9]);
  v1.y = pk2bf(f[10], f[11]);
  v1.z = pk2bf(f[12], f[13]);
  v1.w = pk2bf(f[14], f[15]);
  unsigned short* dst = Wp + ((size_t)((m * 64 + kt) * 8 + nt)) * 512 + lane * 16;
  ((uint4*)dst)[0] = v0;
  ((uint4*)dst)[1] = v1;
}

// ---------------------------------------------------------------------------
// Kernel 1: fused QKV projection, bf16 WMMA, K = C = 2048 in steps of 32.
// 384 threads = 12 waves: wave -> (matrix m = w>>2, column group cg = w&3).
// Each wave computes 32 rows x 32 cols => 2x2 fragments of 16x16.
// The 32x32 fp32 x-tile is DMA'd into LDS with global_load_async_to_lds_b128
// (ASYNCcnt path, §08_async_tensor): waves 0..7 issue one b128 per lane,
// s_wait_asynccnt 0, then barrier.  Each wave converts its A-fragment to
// bf16 in registers with native v_cvt next to the WMMAs.
// Outputs: q bf16 [BT,H] (pre-scaled by H^-0.5 * log2(e)), k bf16 [BT,H],
//          vT bf16 [B][H][T] (transposed so PV B-fragments are contiguous).
// ---------------------------------------------------------------------------
__global__ __launch_bounds__(384)
void qkv_kernel(const float* __restrict__ x,
                const unsigned short* __restrict__ Wp,
                unsigned short* __restrict__ qo,
                unsigned short* __restrict__ ko,
                unsigned short* __restrict__ vto) {
  __shared__ float xtf[32 * 32];   // raw fp32 tile, filled by async DMA

  const int tid  = threadIdx.x;
  const int wave = tid >> 5;
  const int lane = tid & 31;
  const int m    = wave >> 2;     // 0=q 1=k 2=v
  const int cg   = wave & 3;      // column group of 32
  const int rowbase = blockIdx.x * 32;
  const int m16 = lane & 15, kh = lane >> 4;

  // per-lane async-copy coordinates (waves 0..7 move 4 rows each)
  const int srow = wave * 4 + (lane >> 3);
  const int scol = (lane & 7) * 4;
  const unsigned slds = (unsigned)(size_t)&xtf[srow * 32 + scol];

  v8f acc[2][2] = {};

  for (int kc = 0; kc < CC / 32; ++kc) {
    // ---- async DMA x[rowbase..+32, kc*32..+32) fp32 into LDS ----
    if (wave < 8) {
      const float* gp = x + (size_t)(rowbase + srow) * CC + kc * 32 + scol;
      asm volatile("global_load_async_to_lds_b128 %0, %1, off"
                   :: "v"(slds), "v"(gp) : "memory");
      asm volatile("s_wait_asynccnt 0x0" ::: "memory");
    }
    __syncthreads();

    // ---- A fragments: ds_load_b128 of f32 + native cvt to bf16 ----
    FragBF A[2];
#pragma unroll
    for (int mt = 0; mt < 2; ++mt) {
      const float* pA = &xtf[(mt * 16 + m16) * 32 + kh * 8];
      float4 a0 = *(const float4*)pA;
      float4 a1 = *(const float4*)(pA + 4);
      float4 b0 = *(const float4*)(pA + 16);
      float4 b1 = *(const float4*)(pA + 20);
      A[mt].q[0].x = pk2bf(a0.x, a0.y);
      A[mt].q[0].y = pk2bf(a0.z, a0.w);
      A[mt].q[0].z = pk2bf(a1.x, a1.y);
      A[mt].q[0].w = pk2bf(a1.z, a1.w);
      A[mt].q[1].x = pk2bf(b0.x, b0.y);
      A[mt].q[1].y = pk2bf(b0.z, b0.w);
      A[mt].q[1].z = pk2bf(b1.x, b1.y);
      A[mt].q[1].w = pk2bf(b1.z, b1.w);
    }

    // ---- B fragments from packed W, then 4 WMMAs ----
#pragma unroll
    for (int nt = 0; nt < 2; ++nt) {
      FragBF Bf;
      const unsigned short* bp =
          Wp + ((size_t)((m * 64 + kc) * 8 + (cg * 2 + nt))) * 512 + lane * 16;
      Bf.q[0] = ((const uint4*)bp)[0];
      Bf.q[1] = ((const uint4*)bp)[1];
#pragma unroll
      for (int mt = 0; mt < 2; ++mt)
        acc[mt][nt] = __builtin_amdgcn_wmma_f32_16x16x32_bf16(
            false, A[mt].v, false, Bf.v, (short)0, acc[mt][nt], false, false);
    }
    __syncthreads();   // protect LDS tile before next iteration's DMA
  }

  // fold softmax scale AND log2(e) into q so attn uses raw exp2
  const float qscale = 0.08838834764831845f * 1.4426950408889634f;
#pragma unroll
  for (int mt = 0; mt < 2; ++mt)
#pragma unroll
    for (int nt = 0; nt < 2; ++nt)
#pragma unroll
      for (int r = 0; r < 8; ++r) {
        int R   = rowbase + mt * 16 + kh * 8 + r;
        int col = cg * 32 + nt * 16 + m16;
        float f = acc[mt][nt][r];
        if (m == 0) {
          qo[(size_t)R * HH + col] = bf1(f * qscale);
        } else if (m == 1) {
          ko[(size_t)R * HH + col] = bf1(f);
        } else {
          int b = R >> 11, t = R & (TT - 1);
          vto[((size_t)(b * HH + col)) * TT + t] = bf1(f);
        }
      }
}

// ---------------------------------------------------------------------------
// Kernel 2: causal flash attention. 256 threads = 8 waves; wave owns one
// 16-row q block. k processed in 32-wide blocks: 8 WMMAs QK^T + online
// softmax (row stats per-lane, matching D-layout rows) + 8 WMMAs PV.
// P goes D-layout -> A-layout through a per-wave LDS bounce (wave-local,
// synchronized with s_wait_dscnt, NOT a workgroup barrier: trip counts
// diverge across waves).
// ---------------------------------------------------------------------------
__global__ __launch_bounds__(256)
void attn_kernel(const unsigned short* __restrict__ q,
                 const unsigned short* __restrict__ k,
                 const unsigned short* __restrict__ vt,
                 float* __restrict__ out) {
  __shared__ unsigned short plds[8][16 * 32];

  const int tid  = threadIdx.x;
  const int wave = tid >> 5;
  const int lane = tid & 31;
  const int m16 = lane & 15, kh = lane >> 4;
  const int qbt = blockIdx.x * 128 + wave * 16;  // global row base
  const int b   = qbt >> 11;
  const int tq  = qbt & (TT - 1);

  // q A-fragments for all 4 K-chunks of H=128
  FragBF qA[4];
#pragma unroll
  for (int kc = 0; kc < 4; ++kc) {
    const unsigned short* p = q + (size_t)(qbt + m16) * HH + kc * 32 + kh * 8;
    qA[kc].q[0] = *(const uint4*)p;
    qA[kc].q[1] = *(const uint4*)(p + 16);
  }

  v8f acc[8] = {};
  float mrow[8], lrow[8];
#pragma unroll
  for (int r = 0; r < 8; ++r) { mrow[r] = -3.0e38f; lrow[r] = 0.0f; }

  const int nblocks = ((tq + 15) >> 5) + 1;  // last kt0 <= tq, never OOB

  for (int kb = 0; kb < nblocks; ++kb) {
    const int kt0 = kb * 32;

    // ---- S = q . k^T  (scores already in log2 units via q pre-scale) ----
    v8f S0 = {}, S1 = {};
#pragma unroll
    for (int kc = 0; kc < 4; ++kc) {
      FragBF kB0, kB1;
      const unsigned short* p0 =
          k + (size_t)(b * TT + kt0 + m16) * HH + kc * 32 + kh * 8;
      const unsigned short* p1 = p0 + 16 * HH;
      kB0.q[0] = *(const uint4*)p0; kB0.q[1] = *(const uint4*)(p0 + 16);
      kB1.q[0] = *(const uint4*)p1; kB1.q[1] = *(const uint4*)(p1 + 16);
      S0 = __builtin_amdgcn_wmma_f32_16x16x32_bf16(
          false, qA[kc].v, false, kB0.v, (short)0, S0, false, false);
      S1 = __builtin_amdgcn_wmma_f32_16x16x32_bf16(
          false, qA[kc].v, false, kB1.v, (short)0, S1, false, false);
    }

    // ---- causal mask (only diagonal blocks hit this) ----
    if (kt0 + 31 > tq) {
#pragma unroll
      for (int r = 0; r < 8; ++r) {
        int qrow = tq + kh * 8 + r;
        if (kt0 + m16      > qrow) S0[r] = -3.0e38f;
        if (kt0 + 16 + m16 > qrow) S1[r] = -3.0e38f;
      }
    }

    // ---- online softmax: per-row max/sum via width-16 butterflies ----
    float c_r[8];
#pragma unroll
    for (int r = 0; r < 8; ++r) {
      float t = fmaxf(S0[r], S1[r]);
#pragma unroll
      for (int off = 8; off > 0; off >>= 1)
        t = fmaxf(t, __shfl_xor(t, off, 16));
      float mnew = fmaxf(mrow[r], t);
      c_r[r]  = exp2f(mrow[r] - mnew);
      mrow[r] = mnew;
      float p0 = exp2f(S0[r] - mnew);
      float p1 = exp2f(S1[r] - mnew);
      S0[r] = p0; S1[r] = p1;
      float s = p0 + p1;
#pragma unroll
      for (int off = 8; off > 0; off >>= 1)
        s += __shfl_xor(s, off, 16);
      lrow[r] = lrow[r] * c_r[r] + s;
    }

    // ---- rescale output accumulators ----
#pragma unroll
    for (int nh = 0; nh < 8; ++nh)
#pragma unroll
      for (int r = 0; r < 8; ++r)
        acc[nh][r] *= c_r[r];

    // ---- P: D-layout f32 -> bf16 A-fragment via wave-private LDS ----
#pragma unroll
    for (int r = 0; r < 8; ++r) {
      plds[wave][(kh * 8 + r) * 32 + m16]      = bf1(S0[r]);
      plds[wave][(kh * 8 + r) * 32 + 16 + m16] = bf1(S1[r]);
    }
    asm volatile("s_wait_dscnt 0x0" ::: "memory");  // wave-local LDS fence
    FragBF PA;
    {
      const unsigned short* pb = &plds[wave][m16 * 32 + kh * 8];
      PA.q[0] = *(const uint4*)pb;
      PA.q[1] = *(const uint4*)(pb + 16);
    }

    // prefetch next k block while PV math runs
    if (kb + 1 < nblocks)
      __builtin_prefetch(k + (size_t)(b * TT + kt0 + 32 + m16) * HH, 0, 1);

    // ---- acc += P . V  (vT rows give contiguous B-fragments) ----
#pragma unroll
    for (int nh = 0; nh < 8; ++nh) {
      FragBF vB;
      const unsigned short* vp =
          vt + (size_t)(b * HH + nh * 16 + m16) * TT + kt0 + kh * 8;
      vB.q[0] = *(const uint4*)vp;
      vB.q[1] = *(const uint4*)(vp + 16);
      acc[nh] = __builtin_amdgcn_wmma_f32_16x16x32_bf16(
          false, PA.v, false, vB.v, (short)0, acc[nh], false, false);
    }
  }

  // ---- normalize and store fp32 output ----
  float inv[8];
#pragma unroll
  for (int r = 0; r < 8; ++r) inv[r] = 1.0f / lrow[r];
#pragma unroll
  for (int nh = 0; nh < 8; ++nh)
#pragma unroll
    for (int r = 0; r < 8; ++r) {
      int R = qbt + kh * 8 + r;
      out[(size_t)R * HH + nh * 16 + m16] = acc[nh][r] * inv[r];
    }
}

// ---------------------------------------------------------------------------
// Launcher. Workspace layout (bf16 = ushort):
//   q   : BT*H            = 1M elems (2 MB)
//   k   : BT*H            = 1M elems (2 MB)
//   vT  : BT*H            = 1M elems (2 MB)
//   Wp  : 3*64*8*512      = 768K elems (1.5 MB)
// ---------------------------------------------------------------------------
extern "C" void kernel_launch(void* const* d_in, const int* in_sizes, int n_in,
                              void* d_out, int out_size, void* d_ws, size_t ws_size,
                              hipStream_t stream) {
  (void)in_sizes; (void)n_in; (void)out_size; (void)ws_size;
  const float* x  = (const float*)d_in[0];
  const float* Wq = (const float*)d_in[1];
  const float* Wk = (const float*)d_in[2];
  const float* Wv = (const float*)d_in[3];

  unsigned short* qws  = (unsigned short*)d_ws;
  unsigned short* kws  = qws  + (size_t)BT * HH;
  unsigned short* vtws = kws  + (size_t)BT * HH;
  unsigned short* Wp   = vtws + (size_t)BT * HH;

  pack_w_kernel<<<dim3(1536 / 8), dim3(256), 0, stream>>>(Wq, Wk, Wv, Wp);
  qkv_kernel<<<dim3(BT / 32), dim3(384), 0, stream>>>(x, Wp, qws, kws, vtws);
  attn_kernel<<<dim3(BT / 128), dim3(256), 0, stream>>>(qws, kws, vtws,
                                                        (float*)d_out);
}
